// CustomLayer_26749056320330
// MI455X (gfx1250) — matrix operand order repaired and checked by reading, not compiled, not run on hardware
//
#include <hip/hip_runtime.h>
#include <hip/hip_bf16.h>

// ---------------------------------------------------------------------------
// ESN recurrence for MI455X (gfx1250), wave32 + v_wmma_f32_16x16x32_bf16.
//   U = inputs @ W_in + b_in                (memory-bound GEMM, written to d_out)
//   x_t = 0.5*x_{t-1} + 0.5*tanh(U_t + x_{t-1} @ W_res + b_res)   (persistent kernel)
// W_res / W_in pre-transposed+converted to bf16 (B operand: contiguous K per lane).
// State kept as bf16 ping-pong in a WMMA-A-ready row-major (identity) layout so the A
// operand is a single contiguous 32B load per lane (no register shuffles),
// plus fp32 x_{t-1} held in registers for the leak term.
// ---------------------------------------------------------------------------

typedef __attribute__((ext_vector_type(16))) __bf16         v16bf;
typedef __attribute__((ext_vector_type(8)))  float          v8f;
typedef __attribute__((ext_vector_type(16))) unsigned short v16u;

#define B_   128
#define T_   512
#define D_   64
#define N_   1024
#define TN_  (T_ * N_)

// workspace layout (bytes)
#define WRES_T_OFF 0ul                                         // 1024*1024 bf16 = 2 MB
#define WIN_T_OFF  (WRES_T_OFF + (size_t)N_ * N_ * 2)          // 1024*64 bf16
#define XBF_OFF    (WIN_T_OFF  + (size_t)N_ * D_ * 2)          // 2 * 128*1024 bf16 ping-pong
#define SYNC_OFF   (XBF_OFF    + 2ul * (size_t)B_ * N_ * 2)    // 1 uint barrier counter

__device__ __forceinline__ unsigned short f32_to_bf16(float f) {
    unsigned u = __builtin_bit_cast(unsigned, f);
    u += 0x7FFFu + ((u >> 16) & 1u);                 // round-to-nearest-even
    return (unsigned short)(u >> 16);
}

// A-operand layout: identity (contiguous K), so lane `half` reads 16
// contiguous bf16 at offset half*16, matching the contiguous-K B operand.
__device__ __forceinline__ int permA(int n) {
    return n;
}

// --- prep: W_res [K][N] fp32 -> wres_t [N][K] bf16 --------------------------
__global__ void esn_transpose_wres(const float* __restrict__ w,
                                   unsigned short* __restrict__ wt) {
    int idx = blockIdx.x * blockDim.x + threadIdx.x;           // N_*N_ threads
    int n = idx >> 10, k = idx & (N_ - 1);
    wt[(size_t)n * N_ + k] = f32_to_bf16(w[(size_t)k * N_ + n]);
}

// --- prep: W_in [D][N] fp32 -> win_t [N][D] bf16 ----------------------------
__global__ void esn_transpose_win(const float* __restrict__ w,
                                  unsigned short* __restrict__ wt) {
    int idx = blockIdx.x * blockDim.x + threadIdx.x;           // N_*D_ threads
    int n = idx >> 6, d = idx & (D_ - 1);
    wt[(size_t)n * D_ + d] = f32_to_bf16(w[(size_t)d * N_ + n]);
}

// --- phase 1: U = inputs @ W_in + b_in, written to d_out --------------------
// One wave per 16x16 tile of U (M = B*T flattened). K = 64 -> 2 bf16 WMMAs.
__global__ void esn_input_proj(const float* __restrict__ inp,
                               const unsigned short* __restrict__ win_t,
                               const float* __restrict__ b_in,
                               float* __restrict__ out) {
    const int lane   = threadIdx.x & 31;
    const int wid    = threadIdx.x >> 5;
    const int gw     = blockIdx.x * 4 + wid;                   // 0 .. 262143
    const int m_tile = gw >> 6;                                // 4096 M tiles
    const int n_tile = gw & 63;                                // 64 N tiles
    const int half   = lane >> 4;
    const int col    = lane & 15;
    const int n      = n_tile * 16 + col;
    const int row    = m_tile * 16 + col;                      // A-operand row = lane&15

    const float* arow = inp + (size_t)row * D_;
    const float  bias = b_in[n];                               // D column fixed per lane

    v8f acc;
#pragma unroll
    for (int v = 0; v < 8; ++v) acc[v] = bias;

#pragma unroll
    for (int ko = 0; ko < 2; ++ko) {
        const int k_outer = ko * 32;
        v16u au;
#pragma unroll
        for (int g = 0; g < 2; ++g) {                          // K runs: half*16, half*16+8
            const int base = k_outer + half * 16 + g * 8;
#pragma unroll
            for (int i = 0; i < 8; ++i) au[g * 8 + i] = f32_to_bf16(arow[base + i]);
        }
        v16bf a = __builtin_bit_cast(v16bf, au);
        v16u  bu = *(const v16u*)(win_t + (size_t)n * D_ + k_outer + half * 16);
        v16bf b = __builtin_bit_cast(v16bf, bu);
        acc = __builtin_amdgcn_wmma_f32_16x16x32_bf16(false, a, false, b,
                                                      (short)0, acc, false, false);
    }

#pragma unroll
    for (int v = 0; v < 8; ++v) {
        const int m = m_tile * 16 + v + 8 * half;
        out[(size_t)m * N_ + n] = acc[v];
    }
}

// --- x0 = LEAK * tanh(U[:,0]); fp32 back to d_out, row-major bf16 ping -------
__global__ void esn_x0(float* __restrict__ out, unsigned short* __restrict__ xbf0) {
    int idx = blockIdx.x * blockDim.x + threadIdx.x;           // B_*N_ threads
    int b = idx >> 10, n = idx & (N_ - 1);
    size_t o = (size_t)b * TN_ + n;                            // t = 0
    float x = 0.5f * tanhf(out[o]);
    out[o] = x;
    xbf0[(size_t)b * N_ + permA(n)] = f32_to_bf16(x);
}

// --- phase 2: persistent recurrence, grid barrier between steps -------------
// 128 waves, each owns a 32x32 output patch (2x2 WMMA tiles): A reused across
// 2 B columns, B reused across 2 A strips -> 4 WMMAs per 4 loads per K-block.
__global__ void esn_recur(float* __restrict__ out,
                          const unsigned short* __restrict__ wres_t,
                          const float* __restrict__ b_res,
                          unsigned short* __restrict__ xbf,
                          unsigned* __restrict__ sync) {
    const int lane = threadIdx.x & 31;
    const int wid  = threadIdx.x >> 5;
    const int gw   = blockIdx.x * 4 + wid;                     // 0 .. 127
    const int mm   = gw >> 5;                                  // 0..3  (rows mm*32..+31)
    const int nn   = gw & 31;                                  // 0..31 (cols nn*32..+31)
    const int half = lane >> 4;
    const int col  = lane & 15;

    const int rowA0 = mm * 32 + col;                           // A lane rows (M = lane&15)
    const int rowA1 = rowA0 + 16;
    const int nc0   = nn * 32 + col;                           // D/B lane columns
    const int nc1   = nc0 + 16;
    const int pn0   = permA(nc0);                              // bf16 state store columns (identity)
    const int pn1   = permA(nc1);
    const float br0 = b_res[nc0];
    const float br1 = b_res[nc1];
    const unsigned short* bp0 = wres_t + (size_t)nc0 * N_;
    const unsigned short* bp1 = wres_t + (size_t)nc1 * N_;
    const int mb0 = mm * 32;                                   // tile-row bases
    const int mb1 = mm * 32 + 16;

    // x_{t-1} fp32 held in registers; init from x0 in d_out (t = 0)
    v8f xp00, xp01, xp10, xp11;
#pragma unroll
    for (int v = 0; v < 8; ++v) {
        const int m0 = mb0 + v + 8 * half, m1 = mb1 + v + 8 * half;
        xp00[v] = out[(size_t)m0 * TN_ + nc0];
        xp01[v] = out[(size_t)m0 * TN_ + nc1];
        xp10[v] = out[(size_t)m1 * TN_ + nc0];
        xp11[v] = out[(size_t)m1 * TN_ + nc1];
    }

    const unsigned NBLK = gridDim.x;
    unsigned expected = 0;

    for (int t = 1; t < T_; ++t) {
        const unsigned short* xin  = xbf + (size_t)((t - 1) & 1) * (B_ * N_);
        unsigned short*       xout = xbf + (size_t)(t & 1) * (B_ * N_);
        const unsigned short* ap0  = xin + (size_t)rowA0 * N_ + half * 16;
        const unsigned short* ap1  = xin + (size_t)rowA1 * N_ + half * 16;
        const unsigned short* bq0  = bp0 + half * 16;
        const unsigned short* bq1  = bp1 + half * 16;

        v8f c00 = {}, c01 = {}, c10 = {}, c11 = {};
#pragma unroll 2
        for (int kt = 0; kt < 32; ++kt) {
            const int ko = kt * 32;
            v16bf a0 = __builtin_bit_cast(v16bf, *(const v16u*)(ap0 + ko));
            v16bf a1 = __builtin_bit_cast(v16bf, *(const v16u*)(ap1 + ko));
            v16bf b0 = __builtin_bit_cast(v16bf, *(const v16u*)(bq0 + ko));
            v16bf b1 = __builtin_bit_cast(v16bf, *(const v16u*)(bq1 + ko));
            c00 = __builtin_amdgcn_wmma_f32_16x16x32_bf16(false, a0, false, b0, (short)0, c00, false, false);
            c01 = __builtin_amdgcn_wmma_f32_16x16x32_bf16(false, a0, false, b1, (short)0, c01, false, false);
            c10 = __builtin_amdgcn_wmma_f32_16x16x32_bf16(false, a1, false, b0, (short)0, c10, false, false);
            c11 = __builtin_amdgcn_wmma_f32_16x16x32_bf16(false, a1, false, b1, (short)0, c11, false, false);
        }

        // fused leak/tanh epilogue: U_t read from (x_t written to) d_out
#define ESN_EPILOG(ACC, XP, MBASE, NCOL, BR, PN)                               \
        _Pragma("unroll")                                                      \
        for (int v = 0; v < 8; ++v) {                                          \
            const int    m = (MBASE) + v + 8 * half;                           \
            const size_t o = (size_t)m * TN_ + (size_t)t * N_ + (NCOL);        \
            const float  u = out[o];                                           \
            const float  x = 0.5f * (XP)[v] + 0.5f * tanhf((ACC)[v] + u + (BR)); \
            out[o] = x;                                                        \
            (XP)[v] = x;                                                       \
            xout[(size_t)m * N_ + (PN)] = f32_to_bf16(x);                      \
        }
        ESN_EPILOG(c00, xp00, mb0, nc0, br0, pn0)
        ESN_EPILOG(c01, xp01, mb0, nc1, br1, pn1)
        ESN_EPILOG(c10, xp10, mb1, nc0, br0, pn0)
        ESN_EPILOG(c11, xp11, mb1, nc1, br1, pn1)
#undef ESN_EPILOG

        // grid-wide barrier: monotonic counter (no reset races across steps)
        expected += NBLK;
        __syncthreads();
        if (threadIdx.x == 0) {
            __builtin_amdgcn_fence(__ATOMIC_RELEASE, "agent");
            atomicAdd(sync, 1u);
            while (__hip_atomic_load(sync, __ATOMIC_RELAXED,
                                     __HIP_MEMORY_SCOPE_AGENT) < expected) {
                __builtin_amdgcn_s_sleep(1);
            }
        }
        __syncthreads();
        __builtin_amdgcn_fence(__ATOMIC_ACQUIRE, "agent");
    }
}

extern "C" void kernel_launch(void* const* d_in, const int* in_sizes, int n_in,
                              void* d_out, int out_size, void* d_ws, size_t ws_size,
                              hipStream_t stream) {
    (void)in_sizes; (void)n_in; (void)out_size; (void)ws_size;
    const float* inputs = (const float*)d_in[0];   // [B,T,D]
    const float* W_in   = (const float*)d_in[1];   // [D,N]
    const float* b_in   = (const float*)d_in[2];   // [N]
    const float* W_res  = (const float*)d_in[3];   // [N,N]
    const float* b_res  = (const float*)d_in[4];   // [N]
    float* out = (float*)d_out;                    // [B,T,N]

    char* ws = (char*)d_ws;
    unsigned short* wres_t = (unsigned short*)(ws + WRES_T_OFF);
    unsigned short* win_t  = (unsigned short*)(ws + WIN_T_OFF);
    unsigned short* xbf    = (unsigned short*)(ws + XBF_OFF);
    unsigned*       sync   = (unsigned*)(ws + SYNC_OFF);

    hipMemsetAsync(sync, 0, sizeof(unsigned), stream);         // capture-safe

    esn_transpose_wres<<<(N_ * N_) / 256, 256, 0, stream>>>(W_res, wres_t);
    esn_transpose_win <<<(N_ * D_) / 256, 256, 0, stream>>>(W_in, win_t);
    // 4096 M-tiles * 64 N-tiles, 4 waves/block -> 65536 blocks
    esn_input_proj<<<65536, 128, 0, stream>>>(inputs, win_t, b_in, out);
    esn_x0<<<(B_ * N_) / 256, 256, 0, stream>>>(out, xbf);
    // 128 tile-waves (2x2 tiles each), 4 waves/block -> 32 persistent blocks
    esn_recur<<<32, 128, 0, stream>>>(out, wres_t, b_res, xbf, sync);
}
